// BaselineModel_53807350284554
// MI455X (gfx1250) — matrix operand order repaired and checked by reading, not compiled, not run on hardware
//
#include <hip/hip_runtime.h>

// ---- model constants ----
#define Bc 2
#define Sc 2048
#define Dc 1024
#define Hc 16
#define HDc 64
#define Lc 4
#define Mc (Bc * Sc)      // 4096 rows
#define FDc (4 * Dc)      // 4096

#define AS1 __attribute__((address_space(1)))
#define AS3 __attribute__((address_space(3)))

typedef __attribute__((ext_vector_type(16))) __bf16 v16bf;
typedef __attribute__((ext_vector_type(8)))  float  v8f;
typedef __attribute__((ext_vector_type(4)))  int    i4;

union BFrag {
    v16bf v;
    uint4 q[2];
    unsigned short s[16];
};

static __device__ __forceinline__ unsigned short f32_to_bf16(float f) {
    union { float f; unsigned u; } cvt;
    cvt.f = f;
    unsigned r = cvt.u + 0x7FFFu + ((cvt.u >> 16) & 1u);  // round-to-nearest-even
    return (unsigned short)(r >> 16);
}

static __device__ __forceinline__ v8f v8f_zero() {
    v8f z;
#pragma unroll
    for (int i = 0; i < 8; ++i) z[i] = 0.0f;
    return z;
}

// ---- CDNA5 async LDS copy (ASYNCcnt path) ------------------------------
static __device__ __forceinline__ void async_copy_b128(void* lds, const void* g) {
#if __has_builtin(__builtin_amdgcn_global_load_async_to_lds_b128)
    __builtin_amdgcn_global_load_async_to_lds_b128((AS1 i4*)g, (AS3 i4*)lds, 0, 0);
#else
    unsigned lo = (unsigned)(unsigned long long)(AS3 void*)lds;
    asm volatile("global_load_async_to_lds_b128 %0, %1, off"
                 :: "v"(lo), "v"((unsigned long long)g) : "memory");
#endif
}

static __device__ __forceinline__ void wait_asynccnt0() {
#if __has_builtin(__builtin_amdgcn_s_wait_asynccnt)
    __builtin_amdgcn_s_wait_asynccnt(0);
#else
    asm volatile("s_wait_asynccnt 0x0" ::: "memory");
#endif
}

// ---------------------------------------------------------------------------
// Weight convert + transpose: W [L, K, N] f32 (row-major) -> WT [L, N, K] bf16
// ---------------------------------------------------------------------------
__global__ void __launch_bounds__(256)
transpose_to_bf16(const float* __restrict__ W, unsigned short* __restrict__ WT,
                  int K, int N) {
    long long idx = (long long)blockIdx.x * 256 + threadIdx.x;  // over Lmats*N*K
    long long perMat = (long long)N * K;
    int l = (int)(idx / perMat);
    long long r = idx - (long long)l * perMat;
    int n = (int)(r / K);
    int k = (int)(r - (long long)n * K);
    WT[idx] = f32_to_bf16(W[(long long)l * perMat + (long long)k * N + n]);
}

// ---------------------------------------------------------------------------
// RMSNorm: out = w * x * rsqrt(mean(x^2)+eps)  [optional *mul, f32 and/or bf16 out]
// ---------------------------------------------------------------------------
__global__ void __launch_bounds__(256)
rmsnorm_kernel(const float* __restrict__ in, const float* __restrict__ w,
               const float* __restrict__ mul, float* __restrict__ outf,
               unsigned short* __restrict__ outb) {
    __shared__ float red[256];
    int row = blockIdx.x;
    const float* xr = in + (size_t)row * Dc;
    float ssum = 0.f;
    for (int i = threadIdx.x; i < Dc; i += 256) {
        float v = xr[i];
        ssum += v * v;
    }
    red[threadIdx.x] = ssum;
    __syncthreads();
    for (int s = 128; s > 0; s >>= 1) {
        if (threadIdx.x < s) red[threadIdx.x] += red[threadIdx.x + s];
        __syncthreads();
    }
    float inv = rsqrtf(red[0] * (1.0f / Dc) + 1e-6f);
    for (int i = threadIdx.x; i < Dc; i += 256) {
        float v = xr[i] * inv * w[i];
        size_t off = (size_t)row * Dc + i;
        if (mul)  v *= mul[off];
        if (outf) outf[off] = v;
        if (outb) outb[off] = f32_to_bf16(v);
    }
}

// ---------------------------------------------------------------------------
// bf16 WMMA GEMM, LDS-staged + double-buffered via async-to-LDS.
// C[M,N] = A[M,K] * BT[N,K]^T (+bias) (+residual), optional bf16 copy of C.
// Block: 256 thr = 8 waves; tile 64(M)x128(N); k-step 32.
// ---------------------------------------------------------------------------
__global__ void __launch_bounds__(256)
gemm_bf16(const unsigned short* __restrict__ A, const unsigned short* __restrict__ BT,
          const float* __restrict__ bias, const float* __restrict__ residual,
          float* __restrict__ C, unsigned short* __restrict__ Cbf,
          int N, int K) {
    __shared__ __align__(16) unsigned short Ab[2][64 * 32];    //  8 KB
    __shared__ __align__(16) unsigned short Bb[2][128 * 32];   // 16 KB

    int wid  = threadIdx.x >> 5;
    int lane = threadIdx.x & 31;
    int wm = wid & 3, wn = wid >> 2;
    int rowBlk = blockIdx.y * 64;
    int colBlk = blockIdx.x * 128;
    int l16 = lane & 15;
    int hi  = lane >> 4;
    int aRowL = wm * 16 + l16;

    v8f acc[4];
#pragma unroll
    for (int i = 0; i < 4; ++i) acc[i] = v8f_zero();

    // stage one 64x32 A tile + 128x32 B tile per k-step; each wave issues
    // 1 A-chunk + 2 B-chunks of async b128 (32 lanes x 16B each)
    auto issue_tile = [&](int kb, int buf) {
        {
            int row = wid * 8 + (lane >> 2);   // 64 rows over 8 waves
            int ch  = lane & 3;                // 4 x 16B per 64B row
            const unsigned short* g = A + (size_t)(rowBlk + row) * K + kb + ch * 8;
            async_copy_b128(&Ab[buf][row * 32 + ch * 8], g);
        }
#pragma unroll
        for (int t = 0; t < 2; ++t) {
            int col = wid * 16 + t * 8 + (lane >> 2);  // 128 BT rows over 8 waves
            int ch  = lane & 3;
            const unsigned short* g = BT + (size_t)(colBlk + col) * K + kb + ch * 8;
            async_copy_b128(&Bb[buf][col * 32 + ch * 8], g);
        }
    };

    int NK = K >> 5;
    issue_tile(0, 0);

    for (int kt = 0; kt < NK; ++kt) {
        int cur = kt & 1;
        wait_asynccnt0();
        __syncthreads();                         // tile[cur] visible to all waves
        if (kt + 1 < NK) issue_tile((kt + 1) << 5, 1 - cur);

        BFrag a;
        a.q[0] = *(const uint4*)&Ab[cur][aRowL * 32 + hi * 8];
        a.q[1] = *(const uint4*)&Ab[cur][aRowL * 32 + 16 + hi * 8];
        BFrag b[4];
#pragma unroll
        for (int nt = 0; nt < 4; ++nt) {
            int colL = wn * 64 + nt * 16 + l16;
            b[nt].q[0] = *(const uint4*)&Bb[cur][colL * 32 + hi * 8];
            b[nt].q[1] = *(const uint4*)&Bb[cur][colL * 32 + 16 + hi * 8];
        }
#pragma unroll
        for (int nt = 0; nt < 4; ++nt)
            acc[nt] = __builtin_amdgcn_wmma_f32_16x16x32_bf16(
                false, a.v, false, b[nt].v, (short)0, acc[nt], false, false);

        __syncthreads();                         // all reads of tile[cur] done
    }

#pragma unroll
    for (int nt = 0; nt < 4; ++nt) {
        int col = colBlk + wn * 64 + nt * 16 + l16;
        float bs = bias ? bias[col] : 0.f;
#pragma unroll
        for (int j = 0; j < 8; ++j) {
            int row = rowBlk + wm * 16 + j + hi * 8;
            size_t off = (size_t)row * N + col;
            float v = acc[nt][j] + bs;
            if (residual) v += residual[off];
            C[off] = v;
            if (Cbf) Cbf[off] = f32_to_bf16(v);
        }
    }
}

// ---------------------------------------------------------------------------
// uvqk epilogue: u=silu(u) (f32); v->bf16; RoPE(q),RoPE(k)->bf16
// ---------------------------------------------------------------------------
__global__ void __launch_bounds__(256)
post_uvqk(const float* __restrict__ uvqk, const float* __restrict__ td,
          const int* __restrict__ pid, float* __restrict__ u,
          unsigned short* __restrict__ qb, unsigned short* __restrict__ kb_,
          unsigned short* __restrict__ vb) {
    int idx = blockIdx.x * 256 + threadIdx.x;   // M*D total
    int row = idx >> 10;
    int col = idx & (Dc - 1);
    const float* rp = uvqk + (size_t)row * FDc;

    float uu = rp[col];
    u[idx] = uu / (1.f + __expf(-uu));
    vb[idx] = f32_to_bf16(rp[Dc + col]);

    float qv = rp[2 * Dc + col];
    float kv = rp[3 * Dc + col];
    int hd = col & (HDc - 1);
    int fi = hd & 31;
    float pos = (float)pid[row] + 0.1f * __logf(td[row] + 1.f);
    float ang = pos * __expf(-0.28782313662425575f * (float)fi); // 10000^(-fi/32)
    float c = __cosf(ang), sn = __sinf(ang);
    float qr = (hd < 32) ? -rp[2 * Dc + col + 32] : rp[2 * Dc + col - 32];
    float kr = (hd < 32) ? -rp[3 * Dc + col + 32] : rp[3 * Dc + col - 32];
    qb[idx]  = f32_to_bf16(qv * c + qr * sn);
    kb_[idx] = f32_to_bf16(kv * c + kr * sn);
}

// ---------------------------------------------------------------------------
// Fused SiLU-attention: one wave per (b, head, 16-query tile).
// V tile is async-staged to LDS while Q.K^T WMMAs + SiLU run.
// ---------------------------------------------------------------------------
__global__ void __launch_bounds__(32)
attn_kernel(const unsigned short* __restrict__ Q, const unsigned short* __restrict__ Kk,
            const unsigned short* __restrict__ V, float* __restrict__ O) {
    __shared__ __align__(16) unsigned short wt[16 * 32];   // score tile bf16
    __shared__ __align__(16) unsigned short vt[32 * 64];   // V tile (kpos x hd)

    int lane = threadIdx.x;
    int qt = blockIdx.x & (Sc / 16 - 1);
    int bh = blockIdx.x >> 7;                // S/16 = 128
    int b  = bh >> 4;
    int hh = bh & 15;
    int l16 = lane & 15, hi = lane >> 4;
    int qbase = qt * 16;
    size_t rowBase = (size_t)b * Sc;

    const unsigned short* qrow = Q + (rowBase + qbase + l16) * Dc + hh * HDc;
    BFrag aq0, aq1;
    aq0.q[0] = *(const uint4*)(qrow + hi * 8);
    aq0.q[1] = *(const uint4*)(qrow + 16 + hi * 8);
    aq1.q[0] = *(const uint4*)(qrow + 32 + hi * 8);
    aq1.q[1] = *(const uint4*)(qrow + 48 + hi * 8);

    v8f acc[4];
#pragma unroll
    for (int i = 0; i < 4; ++i) acc[i] = v8f_zero();

    const float scale = 0.125f;              // 1/sqrt(64)
    int kendEx = qbase + 16;                 // causal bound (exclusive)

    for (int kb = 0; kb < kendEx; kb += 32) {
        // ---- kick off async V tile [32 x 64] -> LDS (overlaps score WMMAs)
#pragma unroll
        for (int t = 0; t < 8; ++t) {
            int idx = t * 32 + lane;         // 32 rows x 8 chunks of 16B
            int r = idx >> 3;
            int c = idx & 7;
            const unsigned short* g = V + (rowBase + kb + r) * Dc + hh * HDc + c * 8;
            async_copy_b128(&vt[r * 64 + c * 8], g);
        }

        // ---- scores: two 16x16 (q x kpos) tiles, contraction over d
#pragma unroll
        for (int nt = 0; nt < 2; ++nt) {
            const unsigned short* krow =
                Kk + (rowBase + kb + nt * 16 + l16) * Dc + hh * HDc;
            BFrag b0, b1;
            b0.q[0] = *(const uint4*)(krow + hi * 8);
            b0.q[1] = *(const uint4*)(krow + 16 + hi * 8);
            b1.q[0] = *(const uint4*)(krow + 32 + hi * 8);
            b1.q[1] = *(const uint4*)(krow + 48 + hi * 8);
            v8f s = v8f_zero();
            s = __builtin_amdgcn_wmma_f32_16x16x32_bf16(false, aq0.v, false, b0.v,
                                                        (short)0, s, false, false);
            s = __builtin_amdgcn_wmma_f32_16x16x32_bf16(false, aq1.v, false, b1.v,
                                                        (short)0, s, false, false);
            int col = kb + nt * 16 + l16;    // kpos
#pragma unroll
            for (int j = 0; j < 8; ++j) {
                int rowq = qbase + j + hi * 8;
                float x = s[j] * scale;
                float w = (col <= rowq) ? x / (1.f + __expf(-x)) : 0.f;
                wt[(j + hi * 8) * 32 + nt * 16 + l16] = f32_to_bf16(w);
            }
        }

        wait_asynccnt0();                    // V tile landed
        __syncthreads();                     // + wt visible

        BFrag aw;
#pragma unroll
        for (int i = 0; i < 8; ++i) aw.s[i]     = wt[l16 * 32 + hi * 8 + i];
#pragma unroll
        for (int i = 0; i < 8; ++i) aw.s[8 + i] = wt[l16 * 32 + 16 + hi * 8 + i];

#pragma unroll
        for (int nt = 0; nt < 4; ++nt) {
            BFrag bv;
#pragma unroll
            for (int i = 0; i < 8; ++i)
                bv.s[i] = vt[(hi * 8 + i) * 64 + nt * 16 + l16];
#pragma unroll
            for (int i = 0; i < 8; ++i)
                bv.s[8 + i] = vt[(16 + hi * 8 + i) * 64 + nt * 16 + l16];
            acc[nt] = __builtin_amdgcn_wmma_f32_16x16x32_bf16(
                false, aw.v, false, bv.v, (short)0, acc[nt], false, false);
        }
        __syncthreads();
    }

#pragma unroll
    for (int nt = 0; nt < 4; ++nt) {
#pragma unroll
        for (int j = 0; j < 8; ++j) {
            size_t off = (rowBase + qbase + j + hi * 8) * Dc + hh * HDc + nt * 16 + l16;
            O[off] = acc[nt][j];
        }
    }
}

// ---------------------------------------------------------------------------
extern "C" void kernel_launch(void* const* d_in, const int* in_sizes, int n_in,
                              void* d_out, int out_size, void* d_ws, size_t ws_size,
                              hipStream_t stream) {
    const float* x          = (const float*)d_in[0];
    const float* td         = (const float*)d_in[1];
    // d_in[2]: attn_mask — causal tril, computed positionally instead
    const float* uvqk_w     = (const float*)d_in[3];
    const float* uvqk_b     = (const float*)d_in[4];
    const float* gate_w     = (const float*)d_in[5];
    const float* out_w      = (const float*)d_in[6];
    const float* out_b      = (const float*)d_in[7];
    const float* in_norm_w  = (const float*)d_in[8];
    const float* last_norm_w= (const float*)d_in[9];
    const int*   pid        = (const int*)d_in[10];

    size_t off = 0;
    char* base = (char*)d_ws;
    auto carve = [&](size_t bytes) -> void* {
        void* p = base + off;
        off += (bytes + 255) & ~(size_t)255;
        return p;
    };
    unsigned short* wqT  = (unsigned short*)carve((size_t)Lc * FDc * Dc * 2); // 32 MB
    unsigned short* woT  = (unsigned short*)carve((size_t)Lc * Dc * Dc * 2);  //  8 MB
    float*          h    = (float*)carve((size_t)Mc * Dc * 4);                // 16 MB
    unsigned short* hb   = (unsigned short*)carve((size_t)Mc * Dc * 2);       //  8 MB
    float*          uvqk = (float*)carve((size_t)Mc * FDc * 4);               // 64 MB
    float*          u    = (float*)carve((size_t)Mc * Dc * 4);                // 16 MB
    unsigned short* qb   = (unsigned short*)carve((size_t)Mc * Dc * 2);
    unsigned short* kb   = (unsigned short*)carve((size_t)Mc * Dc * 2);
    unsigned short* vb   = (unsigned short*)carve((size_t)Mc * Dc * 2);
    float*          attn = (float*)carve((size_t)Mc * Dc * 4);
    unsigned short* gated= (unsigned short*)carve((size_t)Mc * Dc * 2);

    {
        long long tot = (long long)Lc * FDc * Dc;
        transpose_to_bf16<<<(unsigned)(tot / 256), 256, 0, stream>>>(uvqk_w, wqT, Dc, FDc);
        long long tot2 = (long long)Lc * Dc * Dc;
        transpose_to_bf16<<<(unsigned)(tot2 / 256), 256, 0, stream>>>(out_w, woT, Dc, Dc);
    }

    rmsnorm_kernel<<<Mc, 256, 0, stream>>>(x, in_norm_w, nullptr, h, hb);

    for (int l = 0; l < Lc; ++l) {
        gemm_bf16<<<dim3(FDc / 128, Mc / 64), 256, 0, stream>>>(
            hb, wqT + (size_t)l * FDc * Dc, uvqk_b + (size_t)l * FDc,
            nullptr, uvqk, nullptr, FDc, Dc);

        post_uvqk<<<(Mc * Dc) / 256, 256, 0, stream>>>(uvqk, td, pid, u, qb, kb, vb);

        attn_kernel<<<Bc * Hc * (Sc / 16), 32, 0, stream>>>(qb, kb, vb, attn);

        rmsnorm_kernel<<<Mc, 256, 0, stream>>>(attn, gate_w + (size_t)l * Dc, u,
                                               nullptr, gated);

        gemm_bf16<<<dim3(Dc / 128, Mc / 64), 256, 0, stream>>>(
            gated, woT + (size_t)l * Dc * Dc, out_b + (size_t)l * Dc,
            h, h, hb, Dc, Dc);
    }

    rmsnorm_kernel<<<Mc, 256, 0, stream>>>(h, last_norm_w, nullptr, (float*)d_out,
                                           nullptr);
}
